// RingDilatedAttentionHilbertGPUOptimized_65755949301792
// MI455X (gfx1250) — compile-verified, matrix operand order checked by reading, not run on hardware
//
#include <hip/hip_runtime.h>

typedef __bf16 bf16;
typedef __bf16 v16bf __attribute__((ext_vector_type(16)));
typedef float  v8f   __attribute__((ext_vector_type(8)));
typedef int    v4i_t __attribute__((vector_size(16)));

#define EMBED 1024
#define HEADS 16
#define DH    64
#define BATCH 2
#define SEQ   8192
#define MROWS (BATCH * SEQ)   // 16384

#define AS1 __attribute__((address_space(1)))
#define AS3 __attribute__((address_space(3)))

#if defined(__has_builtin)
#if __has_builtin(__builtin_amdgcn_global_load_async_to_lds_b128)
#define HAVE_ASYNC_LDS 1
#endif
#endif
#ifndef HAVE_ASYNC_LDS
#define HAVE_ASYNC_LDS 0
#endif

// ---------------------------------------------------------------------------
// Fragment helpers (wave32 WMMA bf16 16x16x32)
// A (16x32, 16-bit): lane holds row m = lane&15; k packing per half (hi=lane>>4):
//   elements 0..7  -> k = hi*8 + 0..7 ; elements 8..15 -> k = 16 + hi*8 + 0..7
// B (32x16): lane holds column n = lane&15 with the same k packing (B^T rows).
// C/D (16x16 f32): element r -> row m = r + hi*8, col n = lane&15.
// ---------------------------------------------------------------------------
union FragU { uint4 u[2]; v16bf v; };

__device__ inline v16bf load_frag(const bf16* row, int kb, int hi) {
  FragU t;
  t.u[0] = *(const uint4*)(row + kb + hi * 8);
  t.u[1] = *(const uint4*)(row + kb + 16 + hi * 8);
  return t.v;
}

__device__ inline v8f wmma_bf16(v16bf a, v16bf b, v8f c) {
  return __builtin_amdgcn_wmma_f32_16x16x32_bf16(false, a, false, b, (short)0, c,
                                                 false, false);
}

// copy 16 bytes global -> LDS (async if available)
__device__ inline void cp_lds16(bf16* l, const bf16* g) {
#if HAVE_ASYNC_LDS
  __builtin_amdgcn_global_load_async_to_lds_b128(
      (AS1 v4i_t*)(v4i_t*)(void*)g, (AS3 v4i_t*)(v4i_t*)(void*)l, 0, 0);
#else
  *(uint4*)l = *(const uint4*)g;
#endif
}

__device__ inline void async_wait_all() {
#if HAVE_ASYNC_LDS
#if __has_builtin(__builtin_amdgcn_s_wait_asynccnt)
  __builtin_amdgcn_s_wait_asynccnt(0);
#else
  asm volatile("s_wait_asynccnt 0" ::: "memory");
#endif
#endif
}

// ---------------------------------------------------------------------------
// f32 -> bf16 conversion
// ---------------------------------------------------------------------------
__global__ void cvt_kernel(const float* __restrict__ src, bf16* __restrict__ dst,
                           int n) {
  int i = blockIdx.x * blockDim.x + threadIdx.x;
  int stride = gridDim.x * blockDim.x;
  for (; i < n; i += stride) dst[i] = (bf16)src[i];
}

// ---------------------------------------------------------------------------
// Hilbert permutation: perm[rank(i)] = i for i < L, rank by Hilbert distance
// ---------------------------------------------------------------------------
__device__ inline int hilbert_d(int side, int x, int y) {
  int d = 0;
  for (int s = side >> 1; s > 0; s >>= 1) {
    int rx = (x & s) ? 1 : 0;
    int ry = (y & s) ? 1 : 0;
    d += s * s * ((3 * rx) ^ ry);
    if (ry == 0) {
      if (rx == 1) { x = s - 1 - x; y = s - 1 - y; }
      int t = x; x = y; y = t;
    }
  }
  return d;
}

__global__ void hilbert_kernel(int* __restrict__ perm, int L, int side) {
  int i = blockIdx.x * blockDim.x + threadIdx.x;
  if (i >= L) return;
  int di = hilbert_d(side, i % side, i / side);
  int rank = 0;
  for (int j = 0; j < L; ++j) {
    int dj = hilbert_d(side, j % side, j / side);
    rank += (dj < di) ? 1 : 0;
  }
  perm[rank] = i;
}

// ---------------------------------------------------------------------------
// Shared GEMM core: 256 threads / 8 waves, block tile 128x64, K-chunk 32,
// double-buffered LDS staging for A (rows) and W (rows = output cols).
// Wave w computes rows [mblock + 16w, +16), 64 cols -> c[4] fragments (32 VGPR).
// ---------------------------------------------------------------------------
#define GMT 128
#define GNT 64
#define GKT 32

__device__ inline void gemm_core(const bf16* __restrict__ A,
                                 const bf16* __restrict__ W, int K, int mblock,
                                 int nblock, bf16* ldsA, bf16* ldsW, v8f c[4]) {
  int t = threadIdx.x;
  int lane = t & 31;
  int wave = t >> 5;
  int hi = (lane >> 4) & 1;
  int ln = lane & 15;

  // A staging: 128 rows x 32 elems; 2 threads/row, 32B each (2 x b128)
  int arow_s = t >> 1;
  int acol_s = (t & 1) * 16;
  // W staging: 64 rows x 32 elems; 4 threads/row, 16B each (1 x b128)
  int wrow_s = t >> 2;
  int wcol_s = (t & 3) * 8;

  const bf16* gA = A + (size_t)(mblock + arow_s) * K + acol_s;
  const bf16* gW = W + (size_t)(nblock + wrow_s) * K + wcol_s;
  bf16* sA = ldsA + arow_s * GKT + acol_s;
  bf16* sW = ldsW + wrow_s * GKT + wcol_s;
  const int ABUF = GMT * GKT;
  const int WBUF = GNT * GKT;

  // prefetch chunk 0 into buffer 0
  cp_lds16(sA, gA);
  cp_lds16(sA + 8, gA + 8);
  cp_lds16(sW, gW);
  async_wait_all();
  __syncthreads();

  int nk = K / GKT;
  int p = 0;
  for (int kc = 0; kc < nk; ++kc) {
    if (kc + 1 < nk) {  // prefetch next chunk into other buffer
      const bf16* gA2 = gA + (size_t)(kc + 1) * GKT;
      const bf16* gW2 = gW + (size_t)(kc + 1) * GKT;
      cp_lds16(sA + (p ^ 1) * ABUF, gA2);
      cp_lds16(sA + (p ^ 1) * ABUF + 8, gA2 + 8);
      cp_lds16(sW + (p ^ 1) * WBUF, gW2);
    }
    // compute from buffer p
    const bf16* arow = ldsA + p * ABUF + (size_t)(wave * 16 + ln) * GKT;
    v16bf a = load_frag(arow, 0, hi);
#pragma unroll
    for (int j = 0; j < 4; ++j) {
      const bf16* wrow = ldsW + p * WBUF + (size_t)(j * 16 + ln) * GKT;
      v16bf b = load_frag(wrow, 0, hi);
      c[j] = wmma_bf16(a, b, c[j]);
    }
    async_wait_all();
    __syncthreads();
    p ^= 1;
  }
}

// QKV projection: M=16384, N=3072, K=1024; epilogue scatters to [3][B*H][S][64]
__global__ void __launch_bounds__(256, 1)
qkv_gemm_kernel(const bf16* __restrict__ A, const bf16* __restrict__ W,
                const float* __restrict__ bias, bf16* __restrict__ qkv) {
  __shared__ __align__(16) bf16 ldsA[2 * GMT * GKT];
  __shared__ __align__(16) bf16 ldsW[2 * GNT * GKT];
  int lane = threadIdx.x & 31;
  int wave = threadIdx.x >> 5;
  int hi = (lane >> 4) & 1;
  int ln = lane & 15;
  int mblock = blockIdx.x * GMT;
  int nblock = blockIdx.y * GNT;

  v8f c[4];
#pragma unroll
  for (int j = 0; j < 4; ++j)
#pragma unroll
    for (int e = 0; e < 8; ++e) c[j][e] = 0.0f;

  gemm_core(A, W, EMBED, mblock, nblock, ldsA, ldsW, c);

#pragma unroll
  for (int j = 0; j < 4; ++j) {
    int gn = nblock + j * 16 + ln;
    int which = gn >> 10;  // 0=Q 1=K 2=V
    int rem = gn & 1023;
    int h = rem >> 6;
    int dh = rem & 63;
    float bv = bias[gn];
#pragma unroll
    for (int r = 0; r < 8; ++r) {
      int m = mblock + wave * 16 + r + hi * 8;
      int b_ = m >> 13;
      int s = m & (SEQ - 1);
      size_t off = ((size_t)which * (BATCH * HEADS) + (size_t)b_ * HEADS + h);
      off = (off * SEQ + s) * DH + dh;
      qkv[off] = (bf16)(c[j][r] + bv);
    }
  }
}

// Output projection: M=16384, N=1024, K=1024; f32 row-major output
__global__ void __launch_bounds__(256, 1)
out_gemm_kernel(const bf16* __restrict__ A, const bf16* __restrict__ W,
                const float* __restrict__ bias, float* __restrict__ out) {
  __shared__ __align__(16) bf16 ldsA[2 * GMT * GKT];
  __shared__ __align__(16) bf16 ldsW[2 * GNT * GKT];
  int lane = threadIdx.x & 31;
  int wave = threadIdx.x >> 5;
  int hi = (lane >> 4) & 1;
  int ln = lane & 15;
  int mblock = blockIdx.x * GMT;
  int nblock = blockIdx.y * GNT;

  v8f c[4];
#pragma unroll
  for (int j = 0; j < 4; ++j)
#pragma unroll
    for (int e = 0; e < 8; ++e) c[j][e] = 0.0f;

  gemm_core(A, W, EMBED, mblock, nblock, ldsA, ldsW, c);

#pragma unroll
  for (int j = 0; j < 4; ++j) {
    int gn = nblock + j * 16 + ln;
    float bv = bias[gn];
#pragma unroll
    for (int r = 0; r < 8; ++r) {
      int m = mblock + wave * 16 + r + hi * 8;
      out[(size_t)m * EMBED + gn] = c[j][r] + bv;
    }
  }
}

// ---------------------------------------------------------------------------
// Flash attention over one dilation group: full softmax over g keys.
// Grid: (g/64, rate, B*H). Block: 128 threads = 4 waves; wave = 16-query tile.
// ---------------------------------------------------------------------------
__global__ void attn_kernel(const bf16* __restrict__ qkv,
                            const int* __restrict__ perm,
                            bf16* __restrict__ attn_out,  // [B*S, EMBED] bf16
                            int seg_base, int g, int rate, float scale) {
  __shared__ __align__(16) bf16 ldsK[32 * 64];   // key chunk, row-major
  __shared__ __align__(16) bf16 ldsVt[64 * 32];  // V chunk transposed [dh][key]
  __shared__ __align__(16) bf16 ldsP[4][16 * 32];

  int lane = threadIdx.x & 31;
  int wave = threadIdx.x >> 5;
  int hi = (lane >> 4) & 1;
  int ln = lane & 15;
  int grp = blockIdx.y;
  int bh = blockIdx.z;
  int b = bh / HEADS;
  int h = bh % HEADS;

  size_t planesz = (size_t)BATCH * HEADS * SEQ * DH;
  const bf16* qbase = qkv + (size_t)bh * SEQ * DH;
  const bf16* kbase = qbase + planesz;
  const bf16* vbase = qbase + 2 * planesz;

  int qtile = blockIdx.x * 4 + wave;
  int iq = qtile * 16 + ln;
  int qpos = seg_base + perm[iq * rate + grp];
  const bf16* qrow = qbase + (size_t)qpos * DH;
  v16bf aq0 = load_frag(qrow, 0, hi);
  v16bf aq1 = load_frag(qrow, 32, hi);

  v8f o[4];
  float mrow[8], lrow[8];
#pragma unroll
  for (int dt = 0; dt < 4; ++dt)
#pragma unroll
    for (int e = 0; e < 8; ++e) o[dt][e] = 0.0f;
#pragma unroll
  for (int r = 0; r < 8; ++r) { mrow[r] = -1e30f; lrow[r] = 0.0f; }

  int nchunk = g >> 5;
  for (int ch = 0; ch < nchunk; ++ch) {
    __syncthreads();
    {  // cooperative K/V chunk staging (32 rows x 64 cols; 16 elems/thread)
      int row = threadIdx.x >> 2;
      int part = (threadIdx.x & 3) * 16;
      int slot = ch * 32 + row;
      int pos = seg_base + perm[slot * rate + grp];
      const bf16* ksrc = kbase + (size_t)pos * DH + part;
      bf16* kdst = &ldsK[row * 64 + part];
      cp_lds16(kdst, ksrc);            // async K staging (ASYNCcnt)
      cp_lds16(kdst + 8, ksrc + 8);
      union { uint4 u[2]; bf16 e[16]; } vv;  // V needs transpose: via regs
      const bf16* vsrc = vbase + (size_t)pos * DH + part;
      vv.u[0] = *(const uint4*)(vsrc);
      vv.u[1] = *(const uint4*)(vsrc + 8);
#pragma unroll
      for (int e = 0; e < 16; ++e) ldsVt[(part + e) * 32 + row] = vv.e[e];
    }
    async_wait_all();
    __syncthreads();

    // --- scores: S = Q * K^T (two 16x16 tiles covering 32 keys) ---
    v8f s0, s1;
#pragma unroll
    for (int e = 0; e < 8; ++e) { s0[e] = 0.0f; s1[e] = 0.0f; }
    {
      const bf16* krow0 = &ldsK[(size_t)(0 * 16 + ln) * 64];
      const bf16* krow1 = &ldsK[(size_t)(1 * 16 + ln) * 64];
      s0 = wmma_bf16(aq0, load_frag(krow0, 0, hi), s0);
      s0 = wmma_bf16(aq1, load_frag(krow0, 32, hi), s0);
      s1 = wmma_bf16(aq0, load_frag(krow1, 0, hi), s1);
      s1 = wmma_bf16(aq1, load_frag(krow1, 32, hi), s1);
    }
#pragma unroll
    for (int e = 0; e < 8; ++e) { s0[e] *= scale; s1[e] *= scale; }

    // --- online softmax (element r -> matrix row r + hi*8) ---
    float scl8[8];
#pragma unroll
    for (int r = 0; r < 8; ++r) {
      float mx = fmaxf(s0[r], s1[r]);
#pragma unroll
      for (int off = 1; off < 16; off <<= 1)
        mx = fmaxf(mx, __shfl_xor(mx, off, 32));
      float mnew = fmaxf(mrow[r], mx);
      float sc = __expf(mrow[r] - mnew);
      float p0 = __expf(s0[r] - mnew);
      float p1 = __expf(s1[r] - mnew);
      s0[r] = p0;
      s1[r] = p1;
      float rs = p0 + p1;
#pragma unroll
      for (int off = 1; off < 16; off <<= 1) rs += __shfl_xor(rs, off, 32);
      lrow[r] = lrow[r] * sc + rs;
      mrow[r] = mnew;
      scl8[r] = sc;
    }

    // --- stage P (bf16, row-major 16x32) through LDS into A-fragment ---
    bf16* myP = ldsP[wave];
#pragma unroll
    for (int r = 0; r < 8; ++r) {
      int m = r + hi * 8;
      myP[m * 32 + ln] = (bf16)s0[r];
      myP[m * 32 + 16 + ln] = (bf16)s1[r];
    }

    // --- rescale accumulators, then O += P * V ---
#pragma unroll
    for (int dt = 0; dt < 4; ++dt)
#pragma unroll
      for (int r = 0; r < 8; ++r) o[dt][r] *= scl8[r];

    v16bf ap = load_frag(&myP[ln * 32], 0, hi);
#pragma unroll
    for (int dt = 0; dt < 4; ++dt) {
      v16bf bv = load_frag(&ldsVt[(size_t)(dt * 16 + ln) * 32], 0, hi);
      o[dt] = wmma_bf16(ap, bv, o[dt]);
    }
  }

  // --- epilogue: normalize and scatter rows back to original positions ---
#pragma unroll
  for (int r = 0; r < 8; ++r) {
    int m = r + hi * 8;
    int iqm = qtile * 16 + m;
    int pos = seg_base + perm[iqm * rate + grp];
    size_t orow = ((size_t)b * SEQ + pos) * EMBED + (size_t)h * DH;
    float inv = 1.0f / lrow[r];
#pragma unroll
    for (int dt = 0; dt < 4; ++dt)
      attn_out[orow + dt * 16 + ln] = (bf16)(o[dt][r] * inv);
  }
}

// ---------------------------------------------------------------------------
extern "C" void kernel_launch(void* const* d_in, const int* in_sizes, int n_in,
                              void* d_out, int out_size, void* d_ws,
                              size_t ws_size, hipStream_t stream) {
  (void)in_sizes; (void)n_in; (void)out_size; (void)ws_size;
  const float* x = (const float*)d_in[0];
  const float* w_qkv = (const float*)d_in[1];
  const float* b_qkv = (const float*)d_in[2];
  const float* w_out = (const float*)d_in[3];
  const float* b_out = (const float*)d_in[4];
  float* out = (float*)d_out;

  char* ws = (char*)d_ws;
  size_t off = 0;
  auto alloc = [&](size_t bytes) -> void* {
    void* p = ws + off;
    off += (bytes + 255) & ~(size_t)255;
    return p;
  };
  bf16* x_bf = (bf16*)alloc((size_t)MROWS * EMBED * 2);
  bf16* wqkv_bf = (bf16*)alloc((size_t)3 * EMBED * EMBED * 2);
  bf16* wout_bf = (bf16*)alloc((size_t)EMBED * EMBED * 2);
  bf16* qkv_bf = (bf16*)alloc((size_t)3 * BATCH * HEADS * SEQ * DH * 2);
  bf16* attn_bf = (bf16*)alloc((size_t)MROWS * EMBED * 2);
  int* perm0 = (int*)alloc(1024 * sizeof(int));
  int* perm2 = (int*)alloc(2048 * sizeof(int));
  int* perm3 = (int*)alloc(4096 * sizeof(int));

  // 1) convert inputs to bf16
  cvt_kernel<<<2048, 256, 0, stream>>>(x, x_bf, MROWS * EMBED);
  cvt_kernel<<<2048, 256, 0, stream>>>(w_qkv, wqkv_bf, 3 * EMBED * EMBED);
  cvt_kernel<<<1024, 256, 0, stream>>>(w_out, wout_bf, EMBED * EMBED);

  // 2) Hilbert permutations (L=1024 side=32; L=2048,4096 side=64)
  hilbert_kernel<<<1024 / 128, 128, 0, stream>>>(perm0, 1024, 32);
  hilbert_kernel<<<2048 / 128, 128, 0, stream>>>(perm2, 2048, 64);
  hilbert_kernel<<<4096 / 128, 128, 0, stream>>>(perm3, 4096, 64);

  // 3) QKV projection
  qkv_gemm_kernel<<<dim3(MROWS / GMT, (3 * EMBED) / GNT), 256, 0, stream>>>(
      x_bf, wqkv_bf, b_qkv, qkv_bf);

  // 4) dilated attention per segment
  const float scale = 0.125f;  // 1/sqrt(64)
  struct SegCfg { int base, L, rate; int* perm; };
  SegCfg segs[4] = {{0, 1024, 1, perm0},
                    {1024, 1024, 2, perm0},
                    {2048, 2048, 4, perm2},
                    {4096, 4096, 8, perm3}};
  for (int s = 0; s < 4; ++s) {
    int g = segs[s].L / segs[s].rate;
    dim3 grid(g / 64, segs[s].rate, BATCH * HEADS);
    attn_kernel<<<grid, 128, 0, stream>>>(qkv_bf, segs[s].perm, attn_bf,
                                          segs[s].base, g, segs[s].rate, scale);
  }

  // 5) output projection -> f32 d_out
  out_gemm_kernel<<<dim3(MROWS / GMT, EMBED / GNT), 256, 0, stream>>>(
      attn_bf, wout_bf, b_out, out);
}
